// TransformerBlock_26388279067335
// MI455X (gfx1250) — matrix operand order repaired and checked by reading, not compiled, not run on hardware
//
#include <hip/hip_runtime.h>
#include <hip/hip_bf16.h>

typedef unsigned short u16;
typedef __attribute__((ext_vector_type(16))) __bf16 v16bf;
typedef __attribute__((ext_vector_type(16))) unsigned short v16u;
typedef __attribute__((ext_vector_type(8)))  unsigned short v8u;
typedef __attribute__((ext_vector_type(8)))  float v8f;

#define HWC (512 * 512)

__device__ __forceinline__ u16 f2bf(float f) {
    unsigned int u = __float_as_uint(f);
    u += 0x7FFFu + ((u >> 16) & 1u);
    return (u16)(u >> 16);
}
__device__ __forceinline__ float bf2f(u16 h) {
    return __uint_as_float(((unsigned int)h) << 16);
}
__device__ __forceinline__ float ldv(const float* p) { return *p; }
__device__ __forceinline__ float ldv(const u16* p)   { return bf2f(*p); }
__device__ __forceinline__ void  stv(float* p, float v) { *p = v; }
__device__ __forceinline__ void  stv(u16* p, float v)   { *p = f2bf(v); }

// ---------------------------------------------------------------------------
// Pack fp32 weights (O x I row-major) into zero-padded bf16 (Mpad x Kpad).
// ---------------------------------------------------------------------------
__global__ __launch_bounds__(256) void packw_kernel(
    const float* __restrict__ W, u16* __restrict__ out,
    int M, int K, int Mpad, int Kpad) {
    int idx = blockIdx.x * 256 + threadIdx.x;
    if (idx >= Mpad * Kpad) return;
    int m = idx / Kpad, k = idx % Kpad;
    out[idx] = (m < M && k < K) ? f2bf(W[m * K + k]) : (u16)0;
}

// ---------------------------------------------------------------------------
// LN1: NCHW x -> LN over C -> NHWC bf16 y, plus NHWC f32 copy of x (residual)
// ---------------------------------------------------------------------------
__global__ __launch_bounds__(256) void ln1_kernel(
    const float* __restrict__ x, const float* __restrict__ w,
    const float* __restrict__ b, float* __restrict__ xh,
    u16* __restrict__ yb, int HW) {
    int p = blockIdx.x * 256 + threadIdx.x;
    if (p >= HW) return;
    float v[64];
    float mu = 0.f;
#pragma unroll
    for (int c = 0; c < 64; ++c) { v[c] = x[(size_t)c * HW + p]; mu += v[c]; }
    mu *= (1.f / 64.f);
    float var = 0.f;
#pragma unroll
    for (int c = 0; c < 64; ++c) { float d = v[c] - mu; var += d * d; }
    float inv = rsqrtf(var * (1.f / 64.f) + 1e-5f);
#pragma unroll
    for (int c = 0; c < 64; ++c) {
        float o = (v[c] - mu) * inv * w[c] + b[c];
        xh[(size_t)p * 64 + c] = v[c];
        yb[(size_t)p * 64 + c] = f2bf(o);
    }
}

// ---------------------------------------------------------------------------
// LN2: NHWC f32 -> LN over C -> NHWC bf16
// ---------------------------------------------------------------------------
__global__ __launch_bounds__(256) void ln2_kernel(
    const float* __restrict__ xa, const float* __restrict__ w,
    const float* __restrict__ b, u16* __restrict__ yb, int HW) {
    int p = blockIdx.x * 256 + threadIdx.x;
    if (p >= HW) return;
    float v[64];
    float mu = 0.f;
#pragma unroll
    for (int c = 0; c < 64; ++c) { v[c] = xa[(size_t)p * 64 + c]; mu += v[c]; }
    mu *= (1.f / 64.f);
    float var = 0.f;
#pragma unroll
    for (int c = 0; c < 64; ++c) { float d = v[c] - mu; var += d * d; }
    float inv = rsqrtf(var * (1.f / 64.f) + 1e-5f);
#pragma unroll
    for (int c = 0; c < 64; ++c)
        yb[(size_t)p * 64 + c] = f2bf((v[c] - mu) * inv * w[c] + b[c]);
}

// ---------------------------------------------------------------------------
// 11-tap depthwise pair (NHWC, lane = channel): out ch [0,64) horizontal,
// [64,128) vertical. block (128,2).
// ---------------------------------------------------------------------------
__global__ __launch_bounds__(256) void dw11_kernel(
    const u16* __restrict__ y, const float* __restrict__ ww,
    const float* __restrict__ wb, const float* __restrict__ hw_,
    const float* __restrict__ hb, u16* __restrict__ out) {
    int oc = threadIdx.x;                       // 0..127
    int p  = blockIdx.x * 2 + threadIdx.y;
    int row = p >> 9, col = p & 511;
    float a;
    if (oc < 64) {
        a = wb[oc];
#pragma unroll
        for (int t = 0; t < 11; ++t) {
            int cc = col + t - 5;
            if (cc >= 0 && cc < 512)
                a += ww[oc * 11 + t] * bf2f(y[(size_t)((row << 9) | cc) * 64 + oc]);
        }
    } else {
        int c = oc - 64;
        a = hb[c];
#pragma unroll
        for (int t = 0; t < 11; ++t) {
            int rr = row + t - 5;
            if (rr >= 0 && rr < 512)
                a += hw_[c * 11 + t] * bf2f(y[(size_t)((rr << 9) | col) * 64 + c]);
        }
    }
    out[(size_t)p * 128 + oc] = f2bf(a);
}

// ---------------------------------------------------------------------------
// 3x3 depthwise (NHWC, lane = channel), pad 1. block (C, 256/C or 1).
// ---------------------------------------------------------------------------
template <typename TIN, typename TOUT, bool BIAS, bool SCALE>
__global__ __launch_bounds__(256) void dw3_nhwc_kernel(
    const TIN* __restrict__ in, const float* __restrict__ w,
    const float* __restrict__ b, const float* __restrict__ scale,
    TOUT* __restrict__ out, int H, int W) {
    int C = blockDim.x;
    int c = threadIdx.x;
    int p = blockIdx.x * blockDim.y + threadIdx.y;
    int r = p / W, col = p % W;
    const float* wc = w + c * 9;
    float a = BIAS ? b[c] : 0.f;
#pragma unroll
    for (int dy = -1; dy <= 1; ++dy) {
#pragma unroll
        for (int dx = -1; dx <= 1; ++dx) {
            int rr = r + dy, cc = col + dx;
            if (rr >= 0 && rr < H && cc >= 0 && cc < W)
                a += wc[(dy + 1) * 3 + (dx + 1)] * ldv(in + ((size_t)rr * W + cc) * C + c);
        }
    }
    if (SCALE) a *= scale[c];
    stv(out + (size_t)p * C + c, a);
}

// ---------------------------------------------------------------------------
// WMMA GEMM (NHWC): Y[p, oc] = sum_ic X[p, ic] * W[oc, ic]
//   A = 16 pixels x 32 ic  (contiguous channels -> two b128 loads / lane)
//   B = 32 ic x 16 oc      (= W[oc, k..k+15]   -> one 32B load / lane)
// One wave: 16-pixel strip x OT*16 ocs, A fragment reused OT times.
// mode 0: Y bf16; mode 1: Yf32 = R(NHWC f32) + acc.
// ---------------------------------------------------------------------------
template <int OT>
__global__ __launch_bounds__(256) void gemm_nhwc_kernel(
    const u16* __restrict__ Wp, const u16* __restrict__ X,
    void* __restrict__ Y, const float* __restrict__ R,
    int Moc, int Npix, int Kpad, int ldx, int ldy, int mode) {
    int wave = threadIdx.x >> 5;
    int lane = threadIdx.x & 31;
    int p0  = (blockIdx.x * 8 + wave) * 16;
    int oc0 = blockIdx.y * (OT * 16);
    if (p0 >= Npix) return;

    int lmod = lane & 15;
    int lhi  = lane >> 4;
    const u16* Abase = X + (size_t)(p0 + lmod) * ldx;
    int ocl = oc0 + lmod;

    v8f acc[OT];
#pragma unroll
    for (int ot = 0; ot < OT; ++ot) acc[ot] = v8f{};

    int KT = Kpad >> 5;
    for (int kt = 0; kt < KT; ++kt) {
        // --- A fragment: lanes 0-15 K+0..15, lanes 16-31 K+16..31 ---
        int ka = (kt << 5) + (lhi << 4);
        v8u alo = *(const v8u*)(Abase + ka);
        v8u ahi = *(const v8u*)(Abase + ka + 8);
        v16u au;
#pragma unroll
        for (int j = 0; j < 8; ++j) { au[j] = alo[j]; au[8 + j] = ahi[j]; }
        v16bf afrag = __builtin_bit_cast(v16bf, au);
        // --- B fragments: one contiguous 32B weight-row load per lane ---
        int kb = (kt << 5) + (lhi << 4);
#pragma unroll
        for (int ot = 0; ot < OT; ++ot) {
            v16u bu = *(const v16u*)(Wp + (size_t)(ocl + ot * 16) * Kpad + kb);
            acc[ot] = __builtin_amdgcn_wmma_f32_16x16x32_bf16(
                false, afrag, false, __builtin_bit_cast(v16bf, bu),
                (short)0, acc[ot], false, false);
        }
    }
    // --- store: reg r -> pixel p0 + r + 8*lhi, oc = oc0 + ot*16 + lmod ---
#pragma unroll
    for (int ot = 0; ot < OT; ++ot) {
#pragma unroll
        for (int r = 0; r < 8; ++r) {
            int p  = p0 + r + (lhi << 3);
            int oc = oc0 + ot * 16 + lmod;
            if (oc < Moc) {
                size_t idx = (size_t)p * ldy + oc;
                float v = acc[ot][r];
                if (mode == 0) ((u16*)Y)[idx] = f2bf(v);
                else           ((float*)Y)[idx] = R[idx] + v;
            }
        }
    }
}

// ---------------------------------------------------------------------------
// 8x8 average pool (NHWC, lane = channel): y[512,512,64] -> xd[64,64,64]
// ---------------------------------------------------------------------------
__global__ __launch_bounds__(256) void avgpool_kernel(
    const u16* __restrict__ y, u16* __restrict__ xd) {
    int c  = threadIdx.x;                  // 64
    int pd = blockIdx.x * 4 + threadIdx.y;
    int rd = pd >> 6, cd = pd & 63;
    float s = 0.f;
#pragma unroll
    for (int i = 0; i < 8; ++i)
#pragma unroll
        for (int j = 0; j < 8; ++j)
            s += bf2f(y[(size_t)((rd * 8 + i) * 512 + cd * 8 + j) * 64 + c]);
    xd[(size_t)pd * 64 + c] = f2bf(s * (1.f / 64.f));
}

// ---------------------------------------------------------------------------
// v stats + tiny matvecs (qkvd NHWC [4096][192], v = ch 128..191)
// ---------------------------------------------------------------------------
__global__ __launch_bounds__(64) void vstat_kernel(
    const float* __restrict__ qkvd, const float* __restrict__ w2,
    const float* __restrict__ w3, float* __restrict__ v2) {
    __shared__ float vmax[64], vmean[64];
    int t = threadIdx.x;
    float mx = -3.4e38f, s = 0.f;
    for (int i = 0; i < 4096; ++i) {
        float x = qkvd[(size_t)i * 192 + 128 + t];
        mx = fmaxf(mx, x); s += x;
    }
    vmax[t] = mx; vmean[t] = s * (1.f / 4096.f);
    __syncthreads();
    float a = 0.f;
    for (int c = 0; c < 64; ++c)
        a += w2[t * 64 + c] * vmax[c] + w3[t * 64 + c] * vmean[c];
    v2[t] = a;
}

// ---------------------------------------------------------------------------
// Per-head channel attention (qkvd NHWC): normalized QK^T, softmax, bf16 out.
// ---------------------------------------------------------------------------
__global__ __launch_bounds__(1024) void attn_kernel(
    const float* __restrict__ qkvd, const float* __restrict__ temp,
    u16* __restrict__ attn) {
    int h = blockIdx.x;
    int j = threadIdx.x, i = threadIdx.y;
    int qc = h * 32, kc = 64 + h * 32;
    __shared__ float qn[32], kn[32], rmax[32], rsum[32];
    __shared__ float sm[32][33];
    if (i == 0) {
        float s = 0.f;
        for (int t = 0; t < 4096; ++t) { float v = qkvd[(size_t)t * 192 + qc + j]; s += v * v; }
        qn[j] = fmaxf(sqrtf(s), 1e-12f);
    }
    if (i == 1) {
        float s = 0.f;
        for (int t = 0; t < 4096; ++t) { float v = qkvd[(size_t)t * 192 + kc + j]; s += v * v; }
        kn[j] = fmaxf(sqrtf(s), 1e-12f);
    }
    __syncthreads();
    float d = 0.f;
    for (int t = 0; t < 4096; ++t)
        d += qkvd[(size_t)t * 192 + qc + i] * qkvd[(size_t)t * 192 + kc + j];
    d = d / (qn[i] * kn[j]) * temp[h];
    sm[i][j] = d;
    __syncthreads();
    if (j == 0) {
        float m = -3.4e38f;
        for (int t = 0; t < 32; ++t) m = fmaxf(m, sm[i][t]);
        rmax[i] = m;
    }
    __syncthreads();
    float e = __expf(d - rmax[i]);
    sm[i][j] = e;
    __syncthreads();
    if (j == 0) {
        float s = 0.f;
        for (int t = 0; t < 32; ++t) s += sm[i][t];
        rsum[i] = s;
    }
    __syncthreads();
    attn[(size_t)h * 1024 + i * 32 + j] = f2bf(e / rsum[i]);
}

// ---------------------------------------------------------------------------
// FFN spatial mix (NHWC, lane = channel): x1c = vert11 + horiz11 + biases,
// out = gelu(x1c)*x2 bf16; channels [170,192) zero (K padding). block 192.
// pin row stride 352 (x1 = ch 0..169, x2 = ch 170..339).
// ---------------------------------------------------------------------------
__global__ __launch_bounds__(192) void ffdw_kernel(
    const u16* __restrict__ pin, const float* __restrict__ whv,
    const float* __restrict__ bhv, const float* __restrict__ whw,
    const float* __restrict__ bhw, u16* __restrict__ out) {
    int c = threadIdx.x;                   // 0..191
    int p = blockIdx.x;
    if (c >= 170) { out[(size_t)p * 192 + c] = 0; return; }
    int r = p >> 9, col = p & 511;
    float a = bhv[c] + bhw[c];
#pragma unroll
    for (int t = 0; t < 11; ++t) {         // vertical (fdwh)
        int rr = r + t - 5;
        if (rr >= 0 && rr < 512)
            a += whv[c * 11 + t] * bf2f(pin[(size_t)((rr << 9) | col) * 352 + c]);
    }
#pragma unroll
    for (int t = 0; t < 11; ++t) {         // horizontal (fdww)
        int cc = col + t - 5;
        if (cc >= 0 && cc < 512)
            a += whw[c * 11 + t] * bf2f(pin[(size_t)((r << 9) | cc) * 352 + c]);
    }
    float g  = 0.5f * a * (1.f + erff(a * 0.70710678118f));
    float x2 = bf2f(pin[(size_t)p * 352 + 170 + c]);
    out[(size_t)p * 192 + c] = f2bf(g * x2);
}

// ---------------------------------------------------------------------------
// Final NHWC f32 -> NCHW d_out
// ---------------------------------------------------------------------------
__global__ __launch_bounds__(256) void tr_kernel(
    const float* __restrict__ fn, float* __restrict__ out, int HW) {
    int p = blockIdx.x * 256 + threadIdx.x;
    if (p >= HW) return;
#pragma unroll
    for (int c = 0; c < 64; ++c)
        out[(size_t)c * HW + p] = fn[(size_t)p * 64 + c];
}

// ---------------------------------------------------------------------------
extern "C" void kernel_launch(void* const* d_in, const int* in_sizes, int n_in,
                              void* d_out, int out_size, void* d_ws, size_t ws_size,
                              hipStream_t stream) {
    (void)in_sizes; (void)n_in; (void)out_size; (void)ws_size;
    const float* x       = (const float*)d_in[0];
    const float* ln1_w   = (const float*)d_in[1];
    const float* ln1_b   = (const float*)d_in[2];
    const float* temp    = (const float*)d_in[3];
    const float* dww_w   = (const float*)d_in[4];
    const float* dww_b   = (const float*)d_in[5];
    const float* dwh_w   = (const float*)d_in[6];
    const float* dwh_b   = (const float*)d_in[7];
    const float* dwhw_w  = (const float*)d_in[8];
    const float* dwhw_b  = (const float*)d_in[9];
    const float* conv1_w = (const float*)d_in[10];
    const float* qkv_w   = (const float*)d_in[11];
    const float* qkvdw_w = (const float*)d_in[12];
    const float* conv2_w = (const float*)d_in[13];
    const float* conv3_w = (const float*)d_in[14];
    const float* c4a_w   = (const float*)d_in[15];
    const float* c4b_w   = (const float*)d_in[16];
    const float* projo_w = (const float*)d_in[17];
    const float* ln2_w   = (const float*)d_in[18];
    const float* ln2_b   = (const float*)d_in[19];
    const float* fin_w   = (const float*)d_in[20];
    const float* fdww_w  = (const float*)d_in[21];
    const float* fdww_b  = (const float*)d_in[22];
    const float* fdwh_w  = (const float*)d_in[23];
    const float* fdwh_b  = (const float*)d_in[24];
    const float* fout_w  = (const float*)d_in[25];

    const size_t HW = HWC;
    char* ws = (char*)d_ws;
    // Region A: [x_nhwc f32 | h128 | y_bf | c4a]  -> later pin, then final_nhwc
    float* x_nh  = (float*)(ws + 0);                     // HW*64*4
    u16*   h128  = (u16*)(ws + HW * 256);                // HW*128*2
    u16*   y_bf  = (u16*)(ws + HW * 512);                // HW*64*2
    u16*   c4a   = (u16*)(ws + HW * 640);                // HW*64*2
    u16*   pin   = (u16*)(ws + 0);                       // HW*352*2 <= HW*768  ok
    float* final = (float*)(ws + 0);                     // HW*64*4  (after pin)
    size_t regB = HW * 768;
    // Region B: [t128 (later hl) | vf] -> later ffin (exactly HW*384 bytes)
    u16* t128 = (u16*)(ws + regB);                       // HW*128*2
    u16* hl   = t128;                                    // reuse after t128 dead
    u16* vf   = (u16*)(ws + regB + HW * 256);            // HW*64*2
    u16* ffin = (u16*)(ws + regB);                       // HW*192*2 == HW*384
    size_t regC = regB + HW * 384;
    float* xattn = (float*)(ws + regC);                  // HW*64*4
    size_t regD = regC + HW * 256;
    u16* y2 = (u16*)(ws + regD);                         // HW*64*2
    size_t oS = regD + HW * 128;
    u16*   xd     = (u16*)(ws + oS);                     // 4096*64*2
    u16*   qkv_bf = (u16*)(ws + oS + 524288);            // 4096*192*2
    float* qkvd   = (float*)(ws + oS + 524288 + 1572864);// 4096*192*4
    float* v2vec  = (float*)((char*)qkvd + 3145728);     // 64*4
    u16*   attnb  = (u16*)((char*)v2vec + 256);          // 2*1024*2
    u16* conv1p = (u16*)((char*)attnb + 4096);           // 64x128
    u16* qkvp   = conv1p + 64 * 128;                     // 192x64
    u16* c4ap   = qkvp + 192 * 64;                       // 64x64
    u16* projop = c4ap + 64 * 64;                        // 64x128
    u16* finp   = projop + 64 * 128;                     // 384x64 (pad 340)
    u16* foutp  = finp + 384 * 64;                       // 64x192 (pad 170)

    dim3 blk(256);
    dim3 gFull((HW + 255) / 256);                        // 1024

    // 0) pack GEMM weights -> padded bf16
    packw_kernel<<<dim3(32), blk, 0, stream>>>(conv1_w, conv1p, 64, 128, 64, 128);
    packw_kernel<<<dim3(48), blk, 0, stream>>>(qkv_w,   qkvp,  192,  64, 192,  64);
    packw_kernel<<<dim3(16), blk, 0, stream>>>(c4a_w,   c4ap,   64,  64,  64,  64);
    packw_kernel<<<dim3(32), blk, 0, stream>>>(projo_w, projop, 64, 128,  64, 128);
    packw_kernel<<<dim3(96), blk, 0, stream>>>(fin_w,   finp,  340,  64, 384,  64);
    packw_kernel<<<dim3(48), blk, 0, stream>>>(fout_w,  foutp,  64, 170,  64, 192);

    // 1) LN1: NCHW x -> y_bf (NHWC bf16) + x_nh (NHWC f32 residual copy)
    ln1_kernel<<<gFull, blk, 0, stream>>>(x, ln1_w, ln1_b, x_nh, y_bf, (int)HW);
    // 2) 11-tap depthwise pair -> t128
    dw11_kernel<<<dim3(HW / 2), dim3(128, 2), 0, stream>>>(y_bf, dww_w, dww_b, dwh_w, dwh_b, t128);
    // 3) 3x3 depthwise (C=128, bias) -> h128
    dw3_nhwc_kernel<u16, u16, true, false><<<dim3(HW / 2), dim3(128, 2), 0, stream>>>(
        t128, dwhw_w, dwhw_b, nullptr, h128, 512, 512);
    // 4) conv1 GEMM (oc 64 <- ic 128) -> hl ch [0,64)   (hl overwrites t128)
    gemm_nhwc_kernel<4><<<dim3(2048, 1), blk, 0, stream>>>(
        conv1p, h128, hl, nullptr, 64, (int)HW, 128, 128, 128, 0);
    // 5) 8x8 avg pool -> xd
    avgpool_kernel<<<dim3(1024), dim3(64, 4), 0, stream>>>(y_bf, xd);
    // 6) qkv GEMM (192 <- 64, N=4096) -> qkv_bf
    gemm_nhwc_kernel<4><<<dim3(32, 3), blk, 0, stream>>>(
        qkvp, xd, qkv_bf, nullptr, 192, 4096, 64, 64, 192, 0);
    // 7) qkv 3x3 depthwise (C=192) -> qkvd f32
    dw3_nhwc_kernel<u16, float, false, false><<<dim3(4096), dim3(192, 1), 0, stream>>>(
        qkv_bf, qkvdw_w, nullptr, nullptr, qkvd, 64, 64);
    // 8) v stats -> v2vec
    vstat_kernel<<<dim3(1), dim3(64), 0, stream>>>(qkvd, conv2_w, conv3_w, v2vec);
    // 9) conv4a GEMM (64 <- 64) -> c4a
    gemm_nhwc_kernel<4><<<dim3(2048, 1), blk, 0, stream>>>(
        c4ap, y_bf, c4a, nullptr, 64, (int)HW, 64, 64, 64, 0);
    // 10) conv4b 3x3 depthwise * v2[c] -> vf
    dw3_nhwc_kernel<u16, u16, false, true><<<dim3(HW / 4), dim3(64, 4), 0, stream>>>(
        c4a, c4b_w, nullptr, v2vec, vf, 512, 512);
    // 11) attention matrices -> attnb
    attn_kernel<<<dim3(2), dim3(32, 32), 0, stream>>>(qkvd, temp, attnb);
    // 12) low = vf x attn^T per head -> hl ch [64,128)
    gemm_nhwc_kernel<2><<<dim3(2048, 1), blk, 0, stream>>>(
        attnb, vf, hl + 64, nullptr, 32, (int)HW, 32, 64, 128, 0);
    gemm_nhwc_kernel<2><<<dim3(2048, 1), blk, 0, stream>>>(
        attnb + 1024, vf + 32, hl + 96, nullptr, 32, (int)HW, 32, 64, 128, 0);
    // 13) projo GEMM (64 <- 128) + residual x_nh -> xattn (NHWC f32)
    gemm_nhwc_kernel<4><<<dim3(2048, 1), blk, 0, stream>>>(
        projop, hl, xattn, x_nh, 64, (int)HW, 128, 128, 64, 1);
    // 14) LN2 -> y2
    ln2_kernel<<<gFull, blk, 0, stream>>>(xattn, ln2_w, ln2_b, y2, (int)HW);
    // 15) fin GEMM (340 <- 64, ldy=352) -> pin (region A reuse)
    gemm_nhwc_kernel<4><<<dim3(2048, 6), blk, 0, stream>>>(
        finp, y2, pin, nullptr, 340, (int)HW, 64, 64, 352, 0);
    // 16) FFN 11-tap pair + GELU * gate -> ffin (192 ch, zero-padded)
    ffdw_kernel<<<dim3(HW), dim3(192), 0, stream>>>(pin, fdwh_w, fdwh_b, fdww_w, fdww_b, ffin);
    // 17) fout GEMM (64 <- 192pad) + residual xattn -> final (NHWC f32)
    gemm_nhwc_kernel<4><<<dim3(2048, 1), blk, 0, stream>>>(
        foutp, ffin, final, xattn, 64, (int)HW, 192, 192, 64, 1);
    // 18) NHWC -> NCHW d_out
    tr_kernel<<<gFull, blk, 0, stream>>>(final, (float*)d_out, (int)HW);
}
